// Conv2dMem_16587163697529
// MI455X (gfx1250) — matrix operand order repaired and checked by reading, not compiled, not run on hardware
//
#include <hip/hip_runtime.h>
#include <hip/hip_bf16.h>

// ---------------------------------------------------------------------------
// Quantized 3x3 conv as a block-scaled implicit GEMM on CDNA5 WMMA.
//   M = B*H*W = 50176, N = COUT = 256, K = CIN*9 = 2304.
//   Fake-quant == integer(<=127) x per-block scale; scale blocks of 32 along K
//   map 1:1 onto V_WMMA_F32_16X16X32_F16 (ints exact in f16, partial sums
//   exact in the f32 accumulator), so the GEMM is bit-exact vs the reference.
// Roofline: ~59 GFLOP vs ~100 MB HBM traffic -> memory bound; x (51 MB) and
// quantized weights (1.2 MB) fit in the 192 MB L2, output streamed NT.
// ---------------------------------------------------------------------------

typedef __attribute__((ext_vector_type(16))) _Float16 v16h;
typedef __attribute__((ext_vector_type(8)))  float    v8f;
typedef __attribute__((ext_vector_type(4)))  float    v4f;

#define CB   16
#define CIN  256
#define HW   56
#define L    3136            // 56*56
#define COUT 256
#define CDIM 2304            // CIN*9
#define NKB  72              // CDIM/32
#define MTILES_PER_IMG 196   // L/16

union Frag16 { v16h v; _Float16 e[16]; uint4 q[2]; };
union Frag8f { v8f  v; float    e[8];  v4f f[2]; };

// ---------------------------------------------------------------------------
// Kernel 1: quantize weights.  wmat[c][o] = weight[o*CDIM + c], blocks (32,32).
// Store integer-valued fp16 transposed: wq[o*CDIM + c]; scales swq[cb*8 + ob].
// ---------------------------------------------------------------------------
__global__ void wquant_kernel(const float* __restrict__ weight,
                              _Float16* __restrict__ wq,
                              float* __restrict__ swq) {
    __shared__ float red[128];
    const int blk = blockIdx.x;          // 0..575
    const int cb  = blk >> 3;            // K block (32 rows of wmat)
    const int ob  = blk & 7;             // 32-wide COUT block
    const int tid = threadIdx.x;         // 0..127

    float vals[8];
    float m = 0.0f;
    #pragma unroll
    for (int i = 0; i < 8; ++i) {
        const int e  = tid + i * 128;    // 0..1023
        const int c  = cb * 32 + (e >> 5);
        const int o  = ob * 32 + (e & 31);
        const float v = weight[o * CDIM + c];
        vals[i] = v;
        m = fmaxf(m, fabsf(v));
    }
    red[tid] = m;
    __syncthreads();
    #pragma unroll
    for (int s = 64; s > 0; s >>= 1) {
        if (tid < s) red[tid] = fmaxf(red[tid], red[tid + s]);
        __syncthreads();
    }
    const float mb    = red[0];
    const float scale = (mb > 0.0f) ? mb * (1.0f / 127.0f) : 1.0f;
    const float inv   = 1.0f / scale;
    #pragma unroll
    for (int i = 0; i < 8; ++i) {
        const int e = tid + i * 128;
        const int c = cb * 32 + (e >> 5);
        const int o = ob * 32 + (e & 31);
        wq[o * CDIM + c] = (_Float16)rintf(vals[i] * inv);  // exact int in f16
    }
    if (tid == 0) swq[blk] = scale;
}

// ---------------------------------------------------------------------------
// Kernel 2: fused im2col + activation quant (LDS) + WMMA GEMM + bias + store.
// One workgroup = 16 output pixels (rows); 4 waves x 64 cols = all 256 cols.
// ---------------------------------------------------------------------------
__global__ void __launch_bounds__(128)
conv_wmma_kernel(const float* __restrict__ x,
                 const _Float16* __restrict__ wq,
                 const float* __restrict__ swq,
                 const float* __restrict__ bias,
                 float* __restrict__ out) {
    __shared__ __align__(16) signed char xq8[16 * CDIM];  // 36864 B
    __shared__ __align__(16) float       sxs[NKB * 16];   //  4608 B

    const int blk  = blockIdx.x;                 // 0..3135
    const int bb   = blk / MTILES_PER_IMG;       // image
    const int l0   = (blk % MTILES_PER_IMG) * 16;// first pixel of tile
    const int wv   = threadIdx.x >> 5;           // wave 0..3
    const int lane = threadIdx.x & 31;

    // ---- Phase 1: im2col gather + per-(row, 32-block) quantization --------
    // task T = cb*16 + row; lane j covers k = cb*32 + j.  Two independent
    // tasks per iteration so two gathers are in flight per wave (L2-latency
    // hiding; x fits in the 192 MB L2 so the 9x amplification stays on-chip).
    for (int T0 = wv * 2; T0 < NKB * 16; T0 += 8) {
        #pragma unroll
        for (int u = 0; u < 2; ++u) {
            const int T   = T0 + u;
            const int row = T & 15;
            const int cb  = T >> 4;
            const int c   = cb * 32 + lane;
            const int cin = c / 9;
            const int rem = c - cin * 9;
            const int kh  = rem / 3;
            const int kw  = rem - kh * 3;
            const int l   = l0 + row;
            const int oh  = l / HW;
            const int ow  = l - oh * HW;
            const int ih  = oh + kh - 1;
            const int iw  = ow + kw - 1;
            float v = 0.0f;
            if ((unsigned)ih < (unsigned)HW && (unsigned)iw < (unsigned)HW)
                v = x[(size_t)(bb * CIN + cin) * L + ih * HW + iw];
            float m = fabsf(v);
            #pragma unroll
            for (int mk = 16; mk > 0; mk >>= 1)
                m = fmaxf(m, __shfl_xor(m, mk, 32));
            const float scale = (m > 0.0f) ? m * (1.0f / 127.0f) : 1.0f;
            xq8[row * CDIM + c] = (signed char)rintf(v / scale);
            if (lane == 0) sxs[cb * 16 + row] = scale;
        }
    }
    __syncthreads();

    // ---- Phase 2: WMMA GEMM over 72 K-blocks ------------------------------
    const int n0    = wv * 64;          // wave's first output channel
    const int arow  = lane & 15;        // A-matrix row for this lane
    const int kba   = (lane >> 4) * 16; // f16 A layout: K base per lane half
    const int kbb   = (lane >> 4) * 16; // f16 B layout: K base per lane half
    const int sbase = (lane >> 4) * 8;  // row-scale base per lane half
    const int swi   = (n0 >> 5);        // wave's first 32-wide COUT block

    Frag8f acc[4];
    #pragma unroll
    for (int s = 0; s < 4; ++s)
        #pragma unroll
        for (int j = 0; j < 8; ++j) acc[s].e[j] = 0.0f;

    const v8f czero = {};

    for (int cb = 0; cb < NKB; ++cb) {
        // A fragment: 16 int8 -> f16 (exact). e[0..15]=K kba..kba+15,
        // contiguous, matching the B fragment's K order.
        const signed char* ap = &xq8[arow * CDIM + cb * 32 + kba];
        const uint2 w0 = *(const uint2*)(ap);
        const uint2 w1 = *(const uint2*)(ap + 8);
        Frag16 a;
        #pragma unroll
        for (int k = 0; k < 4; ++k) {
            a.e[k]      = (_Float16)(int)((int)(w0.x << (24 - 8 * k)) >> 24);
            a.e[4 + k]  = (_Float16)(int)((int)(w0.y << (24 - 8 * k)) >> 24);
            a.e[8 + k]  = (_Float16)(int)((int)(w1.x << (24 - 8 * k)) >> 24);
            a.e[12 + k] = (_Float16)(int)((int)(w1.y << (24 - 8 * k)) >> 24);
        }

        // Combined scales: per-row activation scale x per-(cb,ob) weight
        // scale.  The wave's 4 subtiles span exactly two 32-wide COUT blocks.
        const float sw0 = swq[cb * 8 + swi];
        const float sw1 = swq[cb * 8 + swi + 1];
        Frag8f sx, s0, s1;
        const v4f* sp = (const v4f*)&sxs[cb * 16 + sbase];
        sx.f[0] = sp[0];
        sx.f[1] = sp[1];
        #pragma unroll
        for (int j = 0; j < 8; ++j) {
            s0.e[j] = sx.e[j] * sw0;
            s1.e[j] = sx.e[j] * sw1;
        }

        #pragma unroll
        for (int sub = 0; sub < 4; ++sub) {
            const int col = n0 + sub * 16 + (lane & 15);
            const _Float16* bp = wq + (size_t)col * CDIM + cb * 32 + kbb;
            if (cb + 1 < NKB) __builtin_prefetch(bp + 32, 0, 1);
            Frag16 b;
            const uint4* gp = (const uint4*)bp;   // 32 contiguous bytes
            b.q[0] = gp[0];
            b.q[1] = gp[1];

            v8f p = __builtin_amdgcn_wmma_f32_16x16x32_f16(
                false, a.v, false, b.v, (short)0, czero, false, false);

            const Frag8f& sv = (sub < 2) ? s0 : s1;
            #pragma unroll
            for (int j = 0; j < 8; ++j)
                acc[sub].e[j] = fmaf(p[j], sv.e[j], acc[sub].e[j]);
        }
    }

    // ---- Epilogue: bias + NT store (rows contiguous in l per lane) --------
    #pragma unroll
    for (int sub = 0; sub < 4; ++sub) {
        const int col = n0 + sub * 16 + (lane & 15);
        const float bc = bias[col];
        float* op = out + (size_t)(bb * COUT + col) * L + l0 + (lane >> 4) * 8;
        v4f f0, f1;
        f0.x = acc[sub].e[0] + bc; f0.y = acc[sub].e[1] + bc;
        f0.z = acc[sub].e[2] + bc; f0.w = acc[sub].e[3] + bc;
        f1.x = acc[sub].e[4] + bc; f1.y = acc[sub].e[5] + bc;
        f1.z = acc[sub].e[6] + bc; f1.w = acc[sub].e[7] + bc;
        __builtin_nontemporal_store(f0, (v4f*)op);
        __builtin_nontemporal_store(f1, (v4f*)op + 1);
    }
}

// ---------------------------------------------------------------------------
extern "C" void kernel_launch(void* const* d_in, const int* in_sizes, int n_in,
                              void* d_out, int out_size, void* d_ws, size_t ws_size,
                              hipStream_t stream) {
    const float* x      = (const float*)d_in[0];
    const float* weight = (const float*)d_in[1];
    const float* bias   = (const float*)d_in[2];
    float* out          = (float*)d_out;

    // Workspace layout: fp16 quantized weights (transposed) then scales.
    _Float16* wq = (_Float16*)d_ws;                        // 2304*256*2 B
    float* swq   = (float*)((char*)d_ws + (size_t)CDIM * COUT * sizeof(_Float16));

    wquant_kernel<<<576, 128, 0, stream>>>(weight, wq, swq);
    conv_wmma_kernel<<<CB * MTILES_PER_IMG, 128, 0, stream>>>(x, wq, swq, bias, out);
}
